// LatticeEncoderLayer_73263552135765
// MI455X (gfx1250) — matrix operand order repaired and checked by reading, not compiled, not run on hardware
//
#include <hip/hip_runtime.h>
#include <stdint.h>

typedef __attribute__((ext_vector_type(16))) __bf16 v16bf;
typedef __attribute__((ext_vector_type(8)))  float  v8f;

static constexpr int BB = 16, S = 1024, WW = 5, D = 512, F = 2048;
static constexpr int M  = BB * S;          // 16384 rows
static constexpr float LNEPS = 1e-6f;

// ---------------- helpers ----------------
__device__ __forceinline__ uint16_t f32_to_bf16(float f) {
  uint32_t u = __builtin_bit_cast(uint32_t, f);
  u += 0x7FFFu + ((u >> 16) & 1u);   // round-to-nearest-even
  return (uint16_t)(u >> 16);
}

union Frag {
  uint4 u[2];
  v16bf v;
};

// ---------------- weight f32 -> bf16 ----------------
__global__ __launch_bounds__(256) void cvt_bf16_kernel(const float* __restrict__ src,
                                                       uint16_t* __restrict__ dst, int n) {
  int i = blockIdx.x * 256 + threadIdx.x;
  if (i < n) dst[i] = f32_to_bf16(src[i]);
}

// ---------------- fused LN + windowed attention ----------------
// one block (256 threads) per position m; writes cat[m] = bf16([context, q_ln])  [M, 2D]
__global__ __launch_bounds__(256) void attn_ln_kernel(const float* __restrict__ inputs,
                                                      const float* __restrict__ values,
                                                      const float* __restrict__ ln_g,
                                                      const float* __restrict__ ln_b,
                                                      uint16_t* __restrict__ cat) {
  __shared__ float s_red[8];
  __shared__ float s_bcast;
  const int m  = blockIdx.x;
  const int t  = threadIdx.x;
  const int d0 = t * 2;

  auto blockSum = [&](float x) -> float {
    #pragma unroll
    for (int off = 16; off > 0; off >>= 1) x += __shfl_down(x, off, 32);
    const int wid = t >> 5, lane = t & 31;
    if (lane == 0) s_red[wid] = x;
    __syncthreads();
    if (t == 0) {
      float s = 0.f;
      #pragma unroll
      for (int i = 0; i < 8; ++i) s += s_red[i];
      s_bcast = s;
    }
    __syncthreads();
    return s_bcast;
  };

  const float g0 = ln_g[d0],  g1 = ln_g[d0 + 1];
  const float c0b = ln_b[d0], c1b = ln_b[d0 + 1];

  // LN(q)
  const float* qrow = inputs + (size_t)m * D;
  float x0 = qrow[d0], x1 = qrow[d0 + 1];
  float mu  = blockSum(x0 + x1) * (1.0f / D);
  float dx0 = x0 - mu, dx1 = x1 - mu;
  float var = blockSum(dx0 * dx0 + dx1 * dx1) * (1.0f / D);
  float rs  = rsqrtf(var + LNEPS);
  float q0  = dx0 * rs * g0 + c0b;
  float q1  = dx1 * rs * g1 + c1b;

  // LN(v_w) + scores
  float vv[WW][2];
  float sc[WW];
  #pragma unroll
  for (int w = 0; w < WW; ++w) {
    const float* vrow = values + ((size_t)m * WW + w) * D;
    float y0 = vrow[d0], y1 = vrow[d0 + 1];
    float vmu  = blockSum(y0 + y1) * (1.0f / D);
    float e0 = y0 - vmu, e1 = y1 - vmu;
    float vvar = blockSum(e0 * e0 + e1 * e1) * (1.0f / D);
    float vrs  = rsqrtf(vvar + LNEPS);
    float v0 = e0 * vrs * g0 + c0b;
    float v1 = e1 * vrs * g1 + c1b;
    vv[w][0] = v0; vv[w][1] = v1;
    sc[w] = blockSum(q0 * v0 + q1 * v1);
  }

  // softmax over W (replicated on every thread)
  float mx = sc[0];
  #pragma unroll
  for (int w = 1; w < WW; ++w) mx = fmaxf(mx, sc[w]);
  float den = 0.f, ex[WW];
  #pragma unroll
  for (int w = 0; w < WW; ++w) { ex[w] = __expf(sc[w] - mx); den += ex[w]; }
  float inv = 1.0f / den;

  float ctx0 = 0.f, ctx1 = 0.f;
  #pragma unroll
  for (int w = 0; w < WW; ++w) {
    float a = ex[w] * inv;
    ctx0 += a * vv[w][0];
    ctx1 += a * vv[w][1];
  }

  uint16_t* crow = cat + (size_t)m * (2 * D);
  crow[d0]         = f32_to_bf16(ctx0);
  crow[d0 + 1]     = f32_to_bf16(ctx1);
  crow[D + d0]     = f32_to_bf16(q0);
  crow[D + d0 + 1] = f32_to_bf16(q1);
}

// ---------------- row LayerNorm (f32 in, bf16 out) ----------------
__global__ __launch_bounds__(256) void ln_rows_kernel(const float* __restrict__ src,
                                                      const float* __restrict__ g,
                                                      const float* __restrict__ b,
                                                      uint16_t* __restrict__ dst) {
  __shared__ float s_red[8];
  __shared__ float s_bcast;
  const int m  = blockIdx.x;
  const int t  = threadIdx.x;
  const int d0 = t * 2;

  auto blockSum = [&](float x) -> float {
    #pragma unroll
    for (int off = 16; off > 0; off >>= 1) x += __shfl_down(x, off, 32);
    const int wid = t >> 5, lane = t & 31;
    if (lane == 0) s_red[wid] = x;
    __syncthreads();
    if (t == 0) {
      float s = 0.f;
      #pragma unroll
      for (int i = 0; i < 8; ++i) s += s_red[i];
      s_bcast = s;
    }
    __syncthreads();
    return s_bcast;
  };

  const float* row = src + (size_t)m * D;
  float x0 = row[d0], x1 = row[d0 + 1];
  float mu  = blockSum(x0 + x1) * (1.0f / D);
  float dx0 = x0 - mu, dx1 = x1 - mu;
  float var = blockSum(dx0 * dx0 + dx1 * dx1) * (1.0f / D);
  float rs  = rsqrtf(var + LNEPS);
  uint16_t* drow = dst + (size_t)m * D;
  drow[d0]     = f32_to_bf16(dx0 * rs * g[d0]     + b[d0]);
  drow[d0 + 1] = f32_to_bf16(dx1 * rs * g[d0 + 1] + b[d0 + 1]);
}

// ---------------- bf16 WMMA GEMM: C[M,N] = A[M,K] * B[N,K]^T + epilogue ----------------
// Double-buffered LDS; producer uses GLOBAL_LOAD_ASYNC_TO_LDS_B128 (ASYNCcnt-tracked,
// no VGPR staging); tile k+1 streams into buf^1 while tile k is consumed by WMMAs.
// Loop is peeled so the steady state is branch-free (always issue + s_wait_asynccnt 4).
// MODE 0: dstF = tanh(acc) + resid            (GEMM1: linear_out + residual)
// MODE 1: dstH = bf16(relu(acc + bias))       (GEMM2: w1 + b1 + relu)
// MODE 2: dstF = acc + bias + resid           (GEMM3: w2 + b2 + residual)
template <int MODE>
__global__ __launch_bounds__(256)
void gemm_bf16(const uint16_t* __restrict__ A, const uint16_t* __restrict__ Bm,
               int N, int K,
               const float* __restrict__ bias, const float* __restrict__ resid,
               float* __restrict__ dstF, uint16_t* __restrict__ dstH) {
  constexpr int ROWB  = 80;                 // padded LDS row stride (bytes): conflict-free, 16B aligned
  constexpr int TILEB = 128 * ROWB;         // 10240 B per tile
  constexpr int BUFB  = 2 * TILEB;          // A-tile + B-tile per buffer
  __shared__ __align__(16) unsigned char smem[2 * BUFB];   // 40 KB, double buffered

  const int t     = threadIdx.x;
  const int lane  = t & 31;
  const int wid   = t >> 5;
  const int waveM = wid & 3;               // 4 waves along M (32 rows each)
  const int waveN = wid >> 2;              // 2 waves along N (64 cols each)
  const int l     = lane & 15;
  const int hi    = lane >> 4;

  const int rowA0 = blockIdx.x * 128;      // M block
  const int rowB0 = blockIdx.y * 128;      // N block

  v8f acc[2][4];
  #pragma unroll
  for (int mi = 0; mi < 2; ++mi)
    #pragma unroll
    for (int ni = 0; ni < 4; ++ni)
      acc[mi][ni] = (v8f){0.f, 0.f, 0.f, 0.f, 0.f, 0.f, 0.f, 0.f};

  // tile loader mapping: thread t -> tile row t/2, 32-byte half t%2 (two b128 per matrix)
  const int tr = t >> 1;
  const int th = (t & 1) * 32;
  const uint16_t* gA = A  + (size_t)(rowA0 + tr) * K + (t & 1) * 16;
  const uint16_t* gB = Bm + (size_t)(rowB0 + tr) * K + (t & 1) * 16;

  // wave-relative LDS byte address = low 32 bits of the flat address (LDS aperture)
  const unsigned ldsBase = (unsigned)(uintptr_t)(void*)smem;
  const unsigned myAoff  = ldsBase + (unsigned)(tr * ROWB + th);
  const unsigned myBoff  = myAoff + TILEB;

  auto issue_tile = [&](int k0, int bufsel) {
    unsigned la = myAoff + (unsigned)(bufsel * BUFB);
    unsigned lb = myBoff + (unsigned)(bufsel * BUFB);
    const uint16_t* pa = gA + k0;
    const uint16_t* pb = gB + k0;
    asm volatile(
        "global_load_async_to_lds_b128 %0, %2, off\n\t"
        "global_load_async_to_lds_b128 %0, %2, off offset:16\n\t"
        "global_load_async_to_lds_b128 %1, %3, off\n\t"
        "global_load_async_to_lds_b128 %1, %3, off offset:16"
        :: "v"(la), "v"(lb), "v"(pa), "v"(pb)
        : "memory");
  };

  auto compute_tile = [&](int bufsel) {
    const unsigned char* ldsA = smem + bufsel * BUFB;
    const unsigned char* ldsB = ldsA + TILEB;

    // A fragments: 16x32 bf16, lane<16: row l, K 0..15; lane>=16: row l, K 16..31
    Frag af[2];
    #pragma unroll
    for (int mi = 0; mi < 2; ++mi) {
      const unsigned char* p = ldsA + (waveM * 32 + mi * 16 + l) * ROWB;
      af[mi].u[0] = *(const uint4*)(p + hi * 32);
      af[mi].u[1] = *(const uint4*)(p + 16 + hi * 32);
    }
    // B fragments: 32x16 bf16, lane<16: col l, K 0..15; lane>=16: col l, K 16..31
    Frag bf[4];
    #pragma unroll
    for (int ni = 0; ni < 4; ++ni) {
      const unsigned char* p = ldsB + (waveN * 64 + ni * 16 + l) * ROWB + hi * 32;
      bf[ni].u[0] = ((const uint4*)p)[0];
      bf[ni].u[1] = ((const uint4*)p)[1];
    }

    #pragma unroll
    for (int mi = 0; mi < 2; ++mi)
      #pragma unroll
      for (int ni = 0; ni < 4; ++ni)
        acc[mi][ni] = __builtin_amdgcn_wmma_f32_16x16x32_bf16(
            false, af[mi].v, false, bf[ni].v, (short)0, acc[mi][ni], false, false);
  };

  issue_tile(0, 0);
  int buf = 0;

  // steady state: branch-free (always prefetch next tile, retire current with asynccnt<=4)
  for (int k0 = 0; k0 < K - 32; k0 += 32) {
    __syncthreads();                        // WAR: readers of buf^1 (iter k0-32) are done
    issue_tile(k0 + 32, buf ^ 1);           // stream next tile while this one computes
    // per-wave async loads complete in order: >4 outstanding belong to the next tile
    asm volatile("s_wait_asynccnt 0x4" ::: "memory");
    __syncthreads();                        // RAW: all waves' current-buf tiles visible
    compute_tile(buf);
    buf ^= 1;
  }

  // peeled last tile: nothing left to prefetch
  __syncthreads();
  asm volatile("s_wait_asynccnt 0x0" ::: "memory");
  __syncthreads();
  compute_tile(buf);

  // epilogue: C layout — VGPR r, lanes0-15: (M=r, N=lane); lanes16-31: (M=r+8, N=lane-16)
  #pragma unroll
  for (int mi = 0; mi < 2; ++mi) {
    #pragma unroll
    for (int ni = 0; ni < 4; ++ni) {
      const int gmBase = rowA0 + waveM * 32 + mi * 16 + hi * 8;
      const int gn     = rowB0 + waveN * 64 + ni * 16 + l;
      #pragma unroll
      for (int r = 0; r < 8; ++r) {
        const size_t idx = (size_t)(gmBase + r) * N + gn;
        float v = acc[mi][ni][r];
        if (MODE == 0) {
          dstF[idx] = tanhf(v) + resid[idx];
        } else if (MODE == 1) {
          v += bias[gn];
          dstH[idx] = f32_to_bf16(v > 0.f ? v : 0.f);
        } else {
          dstF[idx] = v + bias[gn] + resid[idx];
        }
      }
    }
  }
}

// ---------------- host launcher ----------------
extern "C" void kernel_launch(void* const* d_in, const int* in_sizes, int n_in,
                              void* d_out, int out_size, void* d_ws, size_t ws_size,
                              hipStream_t stream) {
  const float* inputs   = (const float*)d_in[0];
  const float* values   = (const float*)d_in[1];
  const float* ln_g     = (const float*)d_in[2];
  const float* ln_b     = (const float*)d_in[3];
  const float* w_out    = (const float*)d_in[4];
  const float* ffn_ln_g = (const float*)d_in[5];
  const float* ffn_ln_b = (const float*)d_in[6];
  const float* w1       = (const float*)d_in[7];
  const float* b1       = (const float*)d_in[8];
  const float* w2       = (const float*)d_in[9];
  const float* b2       = (const float*)d_in[10];
  float* out = (float*)d_out;

  // workspace layout (bytes)
  char* ws = (char*)d_ws;
  const size_t off_wout  = 0;                                   // [D, 2D]  bf16
  const size_t off_w1    = off_wout  + (size_t)D * 2 * D * 2;   // [F, D]   bf16
  const size_t off_w2    = off_w1    + (size_t)F * D * 2;       // [D, F]   bf16
  const size_t off_cat   = off_w2    + (size_t)D * F * 2;       // [M, 2D]  bf16
  const size_t off_out   = off_cat   + (size_t)M * 2 * D * 2;   // [M, D]   f32
  const size_t off_h     = off_out   + (size_t)M * D * 4;       // [M, D]   bf16
  const size_t off_inter = off_h     + (size_t)M * D * 2;       // [M, F]   bf16

  uint16_t* wout_bf  = (uint16_t*)(ws + off_wout);
  uint16_t* w1_bf    = (uint16_t*)(ws + off_w1);
  uint16_t* w2_bf    = (uint16_t*)(ws + off_w2);
  uint16_t* cat_bf   = (uint16_t*)(ws + off_cat);
  float*    out_f32  = (float*)(ws + off_out);
  uint16_t* h_bf     = (uint16_t*)(ws + off_h);
  uint16_t* inter_bf = (uint16_t*)(ws + off_inter);

  // 1) weights -> bf16 (layouts already [N, K] row-major for all three GEMMs)
  {
    int n = D * 2 * D;
    cvt_bf16_kernel<<<(n + 255) / 256, 256, 0, stream>>>(w_out, wout_bf, n);
    n = F * D;
    cvt_bf16_kernel<<<(n + 255) / 256, 256, 0, stream>>>(w1, w1_bf, n);
    n = D * F;
    cvt_bf16_kernel<<<(n + 255) / 256, 256, 0, stream>>>(w2, w2_bf, n);
  }

  // 2) fused LN + windowed attention -> cat = [context, q_ln] bf16
  attn_ln_kernel<<<M, 256, 0, stream>>>(inputs, values, ln_g, ln_b, cat_bf);

  // 3) out = tanh(cat @ w_out^T) + inputs         [M,1024]x[1024,512]
  gemm_bf16<0><<<dim3(M / 128, D / 128), 256, 0, stream>>>(
      cat_bf, wout_bf, D, 2 * D, nullptr, inputs, out_f32, nullptr);

  // 4) h = LN(out) bf16
  ln_rows_kernel<<<M, 256, 0, stream>>>(out_f32, ffn_ln_g, ffn_ln_b, h_bf);

  // 5) inter = relu(h @ w1^T + b1) bf16           [M,512]x[512,2048]
  gemm_bf16<1><<<dim3(M / 128, F / 128), 256, 0, stream>>>(
      h_bf, w1_bf, F, D, b1, nullptr, nullptr, inter_bf);

  // 6) d_out = inter @ w2^T + b2 + out            [M,2048]x[2048,512]
  gemm_bf16<2><<<dim3(M / 128, D / 128), 256, 0, stream>>>(
      inter_bf, w2_bf, D, F, b2, out_f32, out, nullptr);
}
